// Segmentor3D_62457414418641
// MI455X (gfx1250) — matrix operand order repaired and checked by reading, not compiled, not run on hardware
//
#include <hip/hip_runtime.h>
#include <hip/hip_bf16.h>

// ---------------------------------------------------------------------------
// CDNA5 WMMA (wave32): v_wmma_f32_16x16x32_bf16, D = A(16x32) * B(32x16) + C
// Fragment element mapping (per ISA 7.12.2):
//   A: lane<16 : row=lane,    e0..7 = K0..7,  e8..15 = K16..23
//      lane>=16: row=lane-16, e0..7 = K8..15, e8..15 = K24..31
//   B: lane<16 : col=lane,    e0..7 = K0..7,  e8..15 = K16..23
//      lane>=16: col=lane-16, e0..7 = K8..15, e8..15 = K24..31
// Fragments are kept CONTIGUOUS per lane (32B) so loads are 2x b128.
// ---------------------------------------------------------------------------
typedef __attribute__((ext_vector_type(16))) __bf16 v16bf;
typedef __attribute__((ext_vector_type(8)))  float  v8f;

__device__ inline float gelu_f(float x) {
  const float c = 0.7978845608028654f;            // tanh approx (jax default)
  return 0.5f * x * (1.f + tanhf(c * (x + 0.044715f * x * x * x)));
}
__device__ inline float sigmoid_f(float x) { return 1.f / (1.f + expf(-x)); }

// ---------------------------------------------------------------------------
// f32 -> bf16 elementwise cast (activations)
// ---------------------------------------------------------------------------
__global__ void cast_bf16_k(const float* __restrict__ in, __bf16* __restrict__ out, int n) {
  int i = blockIdx.x * 256 + threadIdx.x;
  if (i < n) out[i] = (__bf16)in[i];
}

// ---------------------------------------------------------------------------
// Pack a row-major f32 B[K][N] into fragment-ready bf16:
//   P[((kc*NB + nb)*32 + lane)*16 + e],  k = kc*32 + (lane>=16 ? 8:0) + (e>=8 ? 8:0) + e,
//   n = nb*16 + (lane&15);  zero-pad k >= K.
// ---------------------------------------------------------------------------
__global__ void pack_b_k(const float* __restrict__ B, __bf16* __restrict__ P,
                         int K, int N) {
  const int NB = N >> 4;
  const int KC = (K + 31) >> 5;
  const int total = KC * NB * 512;
  const int idx = blockIdx.x * 256 + threadIdx.x;
  if (idx >= total) return;
  const int e    = idx & 15;
  const int lane = (idx >> 4) & 31;
  const int t    = idx >> 9;
  const int kc = t / NB, nb = t % NB;
  const int k = kc * 32 + ((lane >= 16) ? 8 : 0) + ((e >= 8) ? 8 : 0) + e;
  const int n = nb * 16 + (lane & 15);
  P[idx] = (k < K) ? (__bf16)B[(long long)k * N + n] : (__bf16)0.0f;
}

// ---------------------------------------------------------------------------
// Implicit-GEMM 3x3x3 'SAME' conv on (32,96,96), Cout=48, WMMA bf16.
// Block: 128 thr = 4 waves; tile = 128 voxels x 48 ch (2 M-tiles per wave,
// 6 WMMAs per K-chunk per wave). K-loop unrolled x2 with explicit ping-pong
// buffers: LDS A double buffer + B register double buffer (no copy chain);
// loads of chunk kc+1 issue under the WMMAs of chunk kc.
// ---------------------------------------------------------------------------
template <int CIN>
__global__ __launch_bounds__(128)
void conv3x3_wmma(const __bf16* __restrict__ X,   // [32*96*96][CIN] bf16
                  const __bf16* __restrict__ Wp,  // packed [KC][3][32][16] bf16
                  const float*  __restrict__ bias,
                  float*  __restrict__ Yf,        // optional f32 out [M][48]
                  __bf16* __restrict__ Yb,        // optional bf16 out [M][48]
                  int act_gelu) {
  constexpr int D = 32, H = 96, W = 96;
  constexpr int M  = D * H * W;
  constexpr int KT = 27 * CIN;                    // CIN % 8 == 0 -> groups of 8 K
  constexpr int KC = (KT + 31) / 32;              // never cross a tap boundary
  __shared__ __attribute__((aligned(16))) __bf16 sA[2][8 * 32 * 16];

  const int tid  = threadIdx.x;
  const int wave = tid >> 5, lane = tid & 31;
  const int m0   = blockIdx.x * 128;

  v8f acc[2][3] = {};

  auto stage = [&](int kc, int buf) {
    const int k0 = kc * 32;
#pragma unroll
    for (int it = 0; it < 4; ++it) {              // 128 rows x 4 k-groups / 128 thr
      const int u = tid + it * 128;
      const int r = u >> 2, g = u & 3;
      const int m  = m0 + r;
      const int kb = k0 + g * 8;                  // 8 contiguous K = 8 contig channels
      uint4 val = {0u, 0u, 0u, 0u};
      if (kb < KT) {
        const int tap = kb / CIN, c = kb % CIN;
        const int dz = tap / 9 - 1, dy = (tap / 3) % 3 - 1, dx = tap % 3 - 1;
        const int z = m / (H * W), y = (m / W) % H, x = m % W;
        const int zz = z + dz, yy = y + dy, xx = x + dx;
        if (zz >= 0 && zz < D && yy >= 0 && yy < H && xx >= 0 && xx < W)
          val = *(const uint4*)(X + (size_t)((zz * H + yy) * W + xx) * CIN + c);
      }
      // scatter group of 8 into fragment layout: one ds_store_b128
      const int tile = r >> 4, row = r & 15;
      const int lt = row + ((g & 1) << 4);        // dest lane
      const int eb = (g >> 1) << 3;               // e base (0 or 8)
      *(uint4*)(&sA[buf][(tile * 32 + lt) * 16 + eb]) = val;
    }
  };
  auto load_b3 = [&](int kc, v16bf* b) {
#pragma unroll
    for (int nb = 0; nb < 3; ++nb)
      b[nb] = *(const v16bf*)(Wp + ((size_t)kc * 3 + nb) * 512 + lane * 16);
  };
  auto compute = [&](int buf, const v16bf* b) {
    const v16bf a0 = *(const v16bf*)(&sA[buf][((wave * 2 + 0) * 32 + lane) * 16]);
    const v16bf a1 = *(const v16bf*)(&sA[buf][((wave * 2 + 1) * 32 + lane) * 16]);
#pragma unroll
    for (int nb = 0; nb < 3; ++nb) {
      acc[0][nb] = __builtin_amdgcn_wmma_f32_16x16x32_bf16(
          false, a0, false, b[nb], (short)0, acc[0][nb], false, false);
      acc[1][nb] = __builtin_amdgcn_wmma_f32_16x16x32_bf16(
          false, a1, false, b[nb], (short)0, acc[1][nb], false, false);
    }
  };

  stage(0, 0);
  v16bf b0[3], b1[3];
  load_b3(0, b0);
  __syncthreads();

  int kc = 0;
  for (;;) {
    // even phase: compute kc from {sA[0], b0}; fill {sA[1], b1} with kc+1
    if (kc + 1 < KC) {
      stage(kc + 1, 1);
      load_b3(kc + 1, b1);
      if (kc + 2 < KC)
        __builtin_prefetch(Wp + (size_t)(kc + 2) * 3 * 512, 0, 1);  // global_prefetch
    }
    compute(0, b0);
    __syncthreads();
    if (++kc >= KC) break;

    // odd phase: compute kc from {sA[1], b1}; fill {sA[0], b0} with kc+1
    if (kc + 1 < KC) {
      stage(kc + 1, 0);
      load_b3(kc + 1, b0);
      if (kc + 2 < KC)
        __builtin_prefetch(Wp + (size_t)(kc + 2) * 3 * 512, 0, 1);
    }
    compute(1, b1);
    __syncthreads();
    if (++kc >= KC) break;
  }

  // epilogue: C layout — lane<16: rows e, lane>=16: rows 8+e; col = lane&15
#pragma unroll
  for (int mt = 0; mt < 2; ++mt) {
#pragma unroll
    for (int nb = 0; nb < 3; ++nb) {
#pragma unroll
      for (int e = 0; e < 8; ++e) {
        const int row = (lane < 16) ? e : (8 + e);
        const int m = m0 + (wave * 2 + mt) * 16 + row;
        const int c = nb * 16 + (lane & 15);
        if (m < M) {
          float v = acc[mt][nb][e] + bias[c];
          if (act_gelu) v = gelu_f(v);
          if (Yf) Yf[(size_t)m * 48 + c] = v;
          if (Yb) Yb[(size_t)m * 48 + c] = (__bf16)v;
        }
      }
    }
  }
}

// ---------------------------------------------------------------------------
// Dense GEMM, M<=64 rows: C[M][N] = A[M][K] @ B[K][N] + bias.
// A double-buffered through LDS in fragment layout; B register ping-pong from
// packed global. Block: 128 thr = 4 waves, tile 64x64, blockIdx.x = N-tile.
// ---------------------------------------------------------------------------
__global__ __launch_bounds__(128)
void gemm64_wmma(const __bf16* __restrict__ A, const __bf16* __restrict__ Bp,
                 const float* __restrict__ bias, float* __restrict__ Cf,
                 __bf16* __restrict__ Cb, int M, int N, int K, int act_gelu) {
  __shared__ __attribute__((aligned(16))) __bf16 sA[2][4 * 32 * 16];
  const int tid = threadIdx.x, wave = tid >> 5, lane = tid & 31;
  const int NB = N >> 4;
  const int nb0 = blockIdx.x * 4;
  const int KC = (K + 31) >> 5;

  v8f acc[4] = {};

  auto stage = [&](int kc, int buf) {
    const int k0 = kc * 32;
#pragma unroll
    for (int it = 0; it < 2; ++it) {              // 64 rows x 4 groups / 128 thr
      const int u = tid + it * 128;
      const int r = u >> 2, g = u & 3;
      const int kb = k0 + g * 8;
      uint4 val = {0u, 0u, 0u, 0u};
      if (r < M && kb + 7 < K)
        val = *(const uint4*)(A + (size_t)r * K + kb);
      const int tile = r >> 4, row = r & 15;
      const int lt = row + ((g & 1) << 4);
      const int eb = (g >> 1) << 3;
      *(uint4*)(&sA[buf][(tile * 32 + lt) * 16 + eb]) = val;
    }
  };
  auto load_b4 = [&](int kc, v16bf* b) {
#pragma unroll
    for (int nb = 0; nb < 4; ++nb)
      b[nb] = *(const v16bf*)(Bp + ((size_t)kc * NB + nb0 + nb) * 512 + lane * 16);
  };
  auto compute = [&](int buf, const v16bf* b) {
    const v16bf a = *(const v16bf*)(&sA[buf][(wave * 32 + lane) * 16]);
#pragma unroll
    for (int nb = 0; nb < 4; ++nb) {
      acc[nb] = __builtin_amdgcn_wmma_f32_16x16x32_bf16(
          false, a, false, b[nb], (short)0, acc[nb], false, false);
    }
  };

  stage(0, 0);
  v16bf b0[4], b1[4];
  load_b4(0, b0);
  __syncthreads();

  int kc = 0;
  for (;;) {
    if (kc + 1 < KC) { stage(kc + 1, 1); load_b4(kc + 1, b1); }
    compute(0, b0);
    __syncthreads();
    if (++kc >= KC) break;
    if (kc + 1 < KC) { stage(kc + 1, 0); load_b4(kc + 1, b0); }
    compute(1, b1);
    __syncthreads();
    if (++kc >= KC) break;
  }

#pragma unroll
  for (int nb = 0; nb < 4; ++nb) {
#pragma unroll
    for (int e = 0; e < 8; ++e) {
      const int m  = wave * 16 + ((lane < 16) ? e : 8 + e);
      const int nn = (nb0 + nb) * 16 + (lane & 15);
      if (m < M && nn < N) {
        float v = acc[nb][e] + bias[nn];
        if (act_gelu) v = gelu_f(v);
        if (Cf) Cf[(long long)m * N + nn] = v;
        if (Cb) Cb[(long long)m * N + nn] = (__bf16)v;
      }
    }
  }
}

// ---------------------------------------------------------------------------
// Gate: mean over 24^3 patch (zero OOB) -> sigmoid(mean @ ca_w + ca_b)
// ---------------------------------------------------------------------------
__global__ __launch_bounds__(64)
void gate_k(const float* __restrict__ Xf, const float* __restrict__ locations,
            const float* __restrict__ ca_w, const float* __restrict__ ca_b,
            float* __restrict__ gate) {
  __shared__ float smean[48];
  const int n = blockIdx.x, c = threadIdx.x;
  const int z0 = (int)(locations[n * 3 + 0] * 0.25f) - 12;
  const int y0 = (int)(locations[n * 3 + 1] * 0.25f) - 12;
  const int x0 = (int)(locations[n * 3 + 2] * 0.25f) - 12;
  if (c < 48) {
    float s = 0.f;
    for (int v = 0; v < 13824; ++v) {
      const int zz = z0 + v / 576, yy = y0 + (v / 24) % 24, xx = x0 + v % 24;
      if (zz >= 0 && zz < 32 && yy >= 0 && yy < 96 && xx >= 0 && xx < 96)
        s += Xf[((zz * 96 + yy) * 96 + xx) * 48 + c];
    }
    smean[c] = s * (1.f / 13824.f);
  }
  __syncthreads();
  if (c < 48) {
    float a = ca_b[c];
    for (int j = 0; j < 48; ++j) a += smean[j] * ca_w[j * 48 + c];
    gate[n * 48 + c] = sigmoid_f(a);
  }
}

// Gated patch materialization: patches[n][v][c] = valid ? x*gate : 0
__global__ void patch_store_k(const float* __restrict__ Xf,
                              const float* __restrict__ locations,
                              const float* __restrict__ gate,
                              float* __restrict__ patches) {
  const int idx = blockIdx.x * 256 + threadIdx.x;       // 64*13824*48
  if (idx >= 64 * 13824 * 48) return;
  const int c = idx % 48;
  const int t = idx / 48;
  const int v = t % 13824;
  const int n = t / 13824;
  const int z0 = (int)(locations[n * 3 + 0] * 0.25f) - 12;
  const int y0 = (int)(locations[n * 3 + 1] * 0.25f) - 12;
  const int x0 = (int)(locations[n * 3 + 2] * 0.25f) - 12;
  const int zz = z0 + v / 576, yy = y0 + (v / 24) % 24, xx = x0 + v % 24;
  float val = 0.f;
  if (zz >= 0 && zz < 32 && yy >= 0 && yy < 96 && xx >= 0 && xx < 96)
    val = Xf[((zz * 96 + yy) * 96 + xx) * 48 + c] * gate[n * 48 + c];
  patches[idx] = val;
}

// inner(12^3) -> 4^3 linear resize is an exact gather at index 3i+1 (patch idx 7+3i)
__global__ void build_sin_k(const float* __restrict__ patches, __bf16* __restrict__ sinB) {
  const int idx = blockIdx.x * 256 + threadIdx.x;       // 64*3072
  if (idx >= 64 * 3072) return;
  const int n = idx / 3072, r = idx % 3072;
  const int i = r / 768, j = (r / 192) % 4, k = (r / 48) % 4, c = r % 48;
  const int v = ((7 + 3 * i) * 24 + (7 + 3 * j)) * 24 + (7 + 3 * k);
  sinB[idx] = (__bf16)patches[(n * 13824 + v) * 48 + c];
}

// LayerNorm over 1024, writes f32 (output 0) + bf16 (pe1 input)
__global__ __launch_bounds__(256)
void layernorm_k(const float* __restrict__ S, const float* __restrict__ g,
                 const float* __restrict__ b, float* __restrict__ outF,
                 __bf16* __restrict__ outB) {
  __shared__ float red[256];
  const int n = blockIdx.x, t = threadIdx.x;
  const float* row = S + n * 1024;
  float s = 0.f;
  for (int i = t; i < 1024; i += 256) s += row[i];
  red[t] = s; __syncthreads();
  for (int o = 128; o > 0; o >>= 1) { if (t < o) red[t] += red[t + o]; __syncthreads(); }
  const float mu = red[0] * (1.f / 1024.f);
  __syncthreads();
  float v = 0.f;
  for (int i = t; i < 1024; i += 256) { const float d = row[i] - mu; v += d * d; }
  red[t] = v; __syncthreads();
  for (int o = 128; o > 0; o >>= 1) { if (t < o) red[t] += red[t + o]; __syncthreads(); }
  const float rstd = rsqrtf(red[0] * (1.f / 1024.f) + 1e-6f);
  for (int i = t; i < 1024; i += 256) {
    const float val = (row[i] - mu) * rstd * g[i] + b[i];
    outF[n * 1024 + i] = val;
    outB[n * 1024 + i] = (__bf16)val;
  }
}

// p(8,8,8,4) -> trilinear 24^3, project 4->48, add to patches, ReLU (in place)
__global__ __launch_bounds__(256)
void posresize_k(const float* __restrict__ P, const float* __restrict__ projW,
                 float* __restrict__ patches) {
  const int idx = blockIdx.x * 256 + threadIdx.x;       // 64*13824
  if (idx >= 64 * 13824) return;
  const int n = idx / 13824, v = idx % 13824;
  const int ic[3] = {v / 576, (v / 24) % 24, v % 24};
  int c0[3], c1[3]; float w1[3];
#pragma unroll
  for (int d = 0; d < 3; ++d) {
    float f = (ic[d] + 0.5f) * (1.f / 3.f) - 0.5f;
    f = fminf(fmaxf(f, 0.f), 7.f);
    c0[d] = (int)f; c1[d] = min(c0[d] + 1, 7); w1[d] = f - (float)c0[d];
  }
  float p4[4];
#pragma unroll
  for (int q = 0; q < 4; ++q) {
    float acc = 0.f;
#pragma unroll
    for (int m = 0; m < 8; ++m) {
      const int zc = (m & 4) ? c1[0] : c0[0];
      const int yc = (m & 2) ? c1[1] : c0[1];
      const int xc = (m & 1) ? c1[2] : c0[2];
      const float w = ((m & 4) ? w1[0] : 1.f - w1[0]) *
                      ((m & 2) ? w1[1] : 1.f - w1[1]) *
                      ((m & 1) ? w1[2] : 1.f - w1[2]);
      acc += w * P[((n * 8 + zc) * 8 + yc) * 8 * 4 + xc * 4 + q];
    }
    p4[q] = acc;
  }
  float* row = patches + (long long)idx * 48;
  for (int c = 0; c < 48; ++c) {
    const float val = row[c] + p4[0] * projW[c] + p4[1] * projW[48 + c] +
                      p4[2] * projW[96 + c] + p4[3] * projW[144 + c];
    row[c] = fmaxf(val, 0.f);
  }
}

// conv_transpose stride 2, k=2, SAME: out[o] = <in[o>>1], w[o&1 per dim]> + b
__global__ void ctrans_k(const float* __restrict__ RP, const float* __restrict__ ctw,
                         const float* __restrict__ ctb, float* __restrict__ L48) {
  const int idx = blockIdx.x * 256 + threadIdx.x;       // 64 * 48^3
  if (idx >= 64 * 110592) return;
  const int n = idx / 110592, o = idx % 110592;
  const int oz = o / 2304, oy = (o / 48) % 48, ox = o % 48;
  const int v = ((oz >> 1) * 24 + (oy >> 1)) * 24 + (ox >> 1);
  const int kd = ((oz & 1) * 2 + (oy & 1)) * 2 + (ox & 1);
  const float* row = RP + ((long long)n * 13824 + v) * 48;
  const float* w = ctw + kd * 48;
  float acc = ctb[0];
  for (int c = 0; c < 48; ++c) acc += row[c] * w[c];
  L48[idx] = acc;
}

// trilinear 48^3 -> 96^3 (factor 2, half-pixel, edge clamp)
__global__ void up96_k(const float* __restrict__ L48, float* __restrict__ out) {
  const int idx = blockIdx.x * 256 + threadIdx.x;       // 64 * 96^3
  if (idx >= 64 * 884736) return;
  const int n = idx / 884736, o = idx % 884736;
  const int ic[3] = {o / 9216, (o / 96) % 96, o % 96};
  int c0[3], c1[3]; float w1[3];
#pragma unroll
  for (int d = 0; d < 3; ++d) {
    float f = (ic[d] + 0.5f) * 0.5f - 0.5f;
    f = fminf(fmaxf(f, 0.f), 47.f);
    c0[d] = (int)f; c1[d] = min(c0[d] + 1, 47); w1[d] = f - (float)c0[d];
  }
  const float* base = L48 + (long long)n * 110592;
  float acc = 0.f;
#pragma unroll
  for (int m = 0; m < 8; ++m) {
    const int zc = (m & 4) ? c1[0] : c0[0];
    const int yc = (m & 2) ? c1[1] : c0[1];
    const int xc = (m & 1) ? c1[2] : c0[2];
    const float w = ((m & 4) ? w1[0] : 1.f - w1[0]) *
                    ((m & 2) ? w1[1] : 1.f - w1[1]) *
                    ((m & 1) ? w1[2] : 1.f - w1[2]);
    acc += w * base[(zc * 48 + yc) * 48 + xc];
  }
  out[idx] = acc;
}

// plocs + mask tail
__global__ void tail_k(const float* __restrict__ locations, float* __restrict__ out) {
  const int n = threadIdx.x;
  if (n >= 64) return;
  const float lz = locations[n * 3 + 0] * 0.25f;
  const float ly = locations[n * 3 + 1] * 0.25f;
  const float lx = locations[n * 3 + 2] * 0.25f;
  const bool mask = lx >= 0.f;
  out[n]        = mask ? (float)(((int)lz - 12) * 4) : -1.f;
  out[64 + n]   = mask ? (float)(((int)ly - 12) * 4) : -1.f;
  out[128 + n]  = mask ? (float)(((int)lx - 12) * 4) : -1.f;
  out[192 + n]  = mask ? 1.f : 0.f;
}

// ---------------------------------------------------------------------------
extern "C" void kernel_launch(void* const* d_in, const int* in_sizes, int n_in,
                              void* d_out, int out_size, void* d_ws, size_t ws_size,
                              hipStream_t stream) {
  const float* feature   = (const float*)d_in[0];
  const float* locations = (const float*)d_in[1];
  const float* conv0_w   = (const float*)d_in[2];
  const float* conv0_b   = (const float*)d_in[3];
  const float* conv1_w   = (const float*)d_in[4];
  const float* conv1_b   = (const float*)d_in[5];
  const float* ca_w      = (const float*)d_in[6];
  const float* ca_b      = (const float*)d_in[7];
  const float* sig_w     = (const float*)d_in[8];
  const float* sig_b     = (const float*)d_in[9];
  const float* ln_scale  = (const float*)d_in[10];
  const float* ln_bias   = (const float*)d_in[11];
  const float* pe1_w     = (const float*)d_in[12];
  const float* pe1_b     = (const float*)d_in[13];
  const float* pe2_w     = (const float*)d_in[14];
  const float* pe2_b     = (const float*)d_in[15];
  const float* pe3_w     = (const float*)d_in[16];
  const float* pe3_b     = (const float*)d_in[17];
  const float* pe_proj_w = (const float*)d_in[18];
  const float* ct_w      = (const float*)d_in[19];
  const float* ct_b      = (const float*)d_in[20];

  const int M = 32 * 96 * 96;           // 294912 voxels

  char* wp = (char*)d_ws;
  auto alloc = [&](size_t bytes) -> void* {
    void* p = (void*)wp;
    wp += (bytes + 255) & ~(size_t)255;
    return p;
  };
  // packed-weight sizes: ceil(K/32)*NB*512 elements
  const size_t w0P_n  = 216 * 3 * 512;            // K=6912,  N=48
  const size_t w1P_n  = 41 * 3 * 512;             // K=1296,  N=48
  const size_t sigP_n = 96 * 64 * 512;            // K=3072,  N=1024
  const size_t pe1P_n = 32 * 64 * 512;            // K=1024,  N=1024
  const size_t pe3P_n = 32 * 128 * 512;           // K=1024,  N=2048

  __bf16* featB = (__bf16*)alloc((size_t)M * 256 * 2);
  __bf16* w0P   = (__bf16*)alloc(w0P_n * 2);
  __bf16* w1P   = (__bf16*)alloc(w1P_n * 2);
  __bf16* y0B   = (__bf16*)alloc((size_t)M * 48 * 2);
  float*  xF    = (float*)alloc((size_t)M * 48 * 4);
  float*  gate  = (float*)alloc((size_t)64 * 48 * 4);
  float*  patches = (float*)alloc((size_t)64 * 13824 * 48 * 4);
  __bf16* sigP  = (__bf16*)alloc(sigP_n * 2);
  __bf16* pe1P  = (__bf16*)alloc(pe1P_n * 2);
  __bf16* pe2P  = (__bf16*)alloc(pe1P_n * 2);
  __bf16* pe3P  = (__bf16*)alloc(pe3P_n * 2);
  __bf16* sinB  = (__bf16*)alloc((size_t)64 * 3072 * 2);
  float*  sF    = (float*)alloc((size_t)64 * 1024 * 4);
  __bf16* psB   = (__bf16*)alloc((size_t)64 * 1024 * 2);
  __bf16* h1B   = (__bf16*)alloc((size_t)64 * 1024 * 2);
  __bf16* h2B   = (__bf16*)alloc((size_t)64 * 1024 * 2);
  float*  pF    = (float*)alloc((size_t)64 * 2048 * 4);
  float*  L48   = (float*)alloc((size_t)64 * 110592 * 4);

  auto cgrid = [](long long n) { return (int)((n + 255) / 256); };

  // activation cast + weight packing for WMMA operands
  cast_bf16_k<<<cgrid((long long)M * 256), 256, 0, stream>>>(feature, featB, M * 256);
  pack_b_k<<<cgrid(w0P_n), 256, 0, stream>>>(conv0_w, w0P, 27 * 256, 48);
  pack_b_k<<<cgrid(w1P_n), 256, 0, stream>>>(conv1_w, w1P, 27 * 48, 48);
  pack_b_k<<<cgrid(sigP_n), 256, 0, stream>>>(sig_w, sigP, 3072, 1024);
  pack_b_k<<<cgrid(pe1P_n), 256, 0, stream>>>(pe1_w, pe1P, 1024, 1024);
  pack_b_k<<<cgrid(pe1P_n), 256, 0, stream>>>(pe2_w, pe2P, 1024, 1024);
  pack_b_k<<<cgrid(pe3P_n), 256, 0, stream>>>(pe3_w, pe3P, 1024, 2048);

  // conv0 (gelu, bf16 out) then conv1 (f32 out)
  conv3x3_wmma<256><<<M / 128, 128, 0, stream>>>(featB, w0P, conv0_b, nullptr, y0B, 1);
  conv3x3_wmma<48><<<M / 128, 128, 0, stream>>>(y0B, w1P, conv1_b, xF, nullptr, 0);

  // patch gating
  gate_k<<<64, 64, 0, stream>>>(xF, locations, ca_w, ca_b, gate);
  patch_store_k<<<cgrid(64LL * 13824 * 48), 256, 0, stream>>>(xF, locations, gate, patches);

  // signature MLP + LayerNorm (output 0 = patch_sigs at d_out[0..65536))
  build_sin_k<<<cgrid(64 * 3072), 256, 0, stream>>>(patches, sinB);
  gemm64_wmma<<<1024 / 64, 128, 0, stream>>>(sinB, sigP, sig_b, sF, nullptr, 64, 1024, 3072, 0);
  layernorm_k<<<64, 256, 0, stream>>>(sF, ln_scale, ln_bias, (float*)d_out, psB);

  // positional-embedding MLP stack
  gemm64_wmma<<<1024 / 64, 128, 0, stream>>>(psB, pe1P, pe1_b, nullptr, h1B, 64, 1024, 1024, 1);
  gemm64_wmma<<<1024 / 64, 128, 0, stream>>>(h1B, pe2P, pe2_b, nullptr, h2B, 64, 1024, 1024, 1);
  gemm64_wmma<<<2048 / 64, 128, 0, stream>>>(h2B, pe3P, pe3_b, pF, nullptr, 64, 2048, 1024, 0);

  // upsample pos-embed, project, add+ReLU (in-place on patches)
  posresize_k<<<cgrid(64 * 13824), 256, 0, stream>>>(pF, pe_proj_w, patches);

  // transpose conv to 48^3, then trilinear to 96^3 (output 1 = logits)
  ctrans_k<<<cgrid(64 * 110592), 256, 0, stream>>>(patches, ct_w, ct_b, L48);
  up96_k<<<cgrid(64LL * 884736), 256, 0, stream>>>(L48, (float*)d_out + 65536);

  // plocs + mask (outputs 2..5)
  tail_k<<<1, 64, 0, stream>>>(locations, (float*)d_out + 65536 + 64 * 884736);
}
